// MultiAttentionHead_9964324127312
// MI455X (gfx1250) — matrix operand order repaired and checked by reading, not compiled, not run on hardware
//
#include <hip/hip_runtime.h>
#include <hip/hip_bf16.h>

typedef __attribute__((ext_vector_type(16))) __bf16 v16bf;
typedef __attribute__((ext_vector_type(8)))  __bf16 v8bf;
typedef __attribute__((ext_vector_type(4)))  __bf16 v4bf;
typedef __attribute__((ext_vector_type(8)))  float  v8f;
typedef __attribute__((ext_vector_type(4)))  float  v4f;

#define BATCH 4
#define SEQ   8192
#define CDIM  1024
#define ADIM  1024
#define NHEAD 16
#define HDIM  64
#define MROWS (BATCH * SEQ)

#define BM 128
#define BN 128
#define BK 64
#define LDK 72   // bf16 stride: 144 bytes/row, 16B aligned, conflict-free

// ---------------------------------------------------------------------------
// Projection GEMM: out[m, a] = bf16( sum_c X[m,c] * W[a,c] + bias[a] )
// X: [MROWS, CDIM] f32 row-major, W: [ADIM, CDIM] f32 row-major (B^T layout,
// reduction contiguous for both operands -> identical staging pattern).
// 8 waves per WG; wave tile 32x64 = 2x4 WMMA 16x16 tiles, f32 accum.
// ---------------------------------------------------------------------------
__global__ __launch_bounds__(256) void proj_gemm_bf16(const float* __restrict__ X,
                                                      const float* __restrict__ Wm,
                                                      const float* __restrict__ bias,
                                                      __bf16* __restrict__ out) {
  __shared__ __bf16 xs[BM * LDK];
  __shared__ __bf16 wsm[BN * LDK];

  const int tile_n = blockIdx.x * BN;
  const int tile_m = blockIdx.y * BM;
  const int tid  = threadIdx.x;
  const int wave = tid >> 5;
  const int lane = tid & 31;
  const int l16  = lane & 15;
  const int lh   = lane >> 4;
  const int wm   = (wave >> 1) * 32;   // wave row offset: 0,32,64,96
  const int wn   = (wave & 1) * 64;    // wave col offset: 0,64

  const int row0 = tid >> 4;           // 0..15
  const int col4 = (tid & 15) * 4;     // 0..60 step 4

  v8f acc[2][4] = {};

  for (int kb = 0; kb < CDIM; kb += BK) {
    // Stage 128x64 f32 tiles -> bf16 LDS (8 float4 per thread per matrix)
#pragma unroll
    for (int i = 0; i < 8; i++) {
      const int r = row0 + i * 16;
      v4f dx = *(const v4f*)(X  + (size_t)(tile_m + r) * CDIM + kb + col4);
      v4f dw = *(const v4f*)(Wm + (size_t)(tile_n + r) * CDIM + kb + col4);
      v4bf cx, cw;
#pragma unroll
      for (int j = 0; j < 4; j++) { cx[j] = (__bf16)dx[j]; cw[j] = (__bf16)dw[j]; }
      *(v4bf*)(&xs[r * LDK + col4])  = cx;
      *(v4bf*)(&wsm[r * LDK + col4]) = cw;
    }
    __syncthreads();

    if (kb + BK < CDIM) {  // pull next k-tile toward the caches (global_prefetch_b8)
      __builtin_prefetch(X  + (size_t)(tile_m + row0) * CDIM + kb + BK + col4, 0, 1);
      __builtin_prefetch(Wm + (size_t)(tile_n + row0) * CDIM + kb + BK + col4, 0, 1);
    }

#pragma unroll
    for (int kk = 0; kk < BK; kk += 32) {
      v16bf afr[2], bfr[4];
      // A fragment (16x32): lane holds K chunks [8*lh,8*lh+8) and +16  (ISA 7.12.2)
#pragma unroll
      for (int mt = 0; mt < 2; mt++) {
        const __bf16* p = &xs[(wm + mt * 16 + l16) * LDK + kk + lh * 8];
        v8bf lo = *(const v8bf*)p;
        v8bf hi = *(const v8bf*)(p + 16);
        v16bf f;
#pragma unroll
        for (int e = 0; e < 8; e++) { f[e] = lo[e]; f[e + 8] = hi[e]; }
        afr[mt] = f;
      }
      // B fragment (32x16): lane = column, K chunks [8*lh,8*lh+8) and +16 (same as A)
#pragma unroll
      for (int nt = 0; nt < 4; nt++) {
        const __bf16* p = &wsm[(wn + nt * 16 + l16) * LDK + kk + lh * 8];
        v8bf lo = *(const v8bf*)p;
        v8bf hi = *(const v8bf*)(p + 16);
        v16bf f;
#pragma unroll
        for (int e = 0; e < 8; e++) { f[e] = lo[e]; f[e + 8] = hi[e]; }
        bfr[nt] = f;
      }
#pragma unroll
      for (int mt = 0; mt < 2; mt++)
#pragma unroll
        for (int nt = 0; nt < 4; nt++)
          acc[mt][nt] = __builtin_amdgcn_wmma_f32_16x16x32_bf16(
              false, afr[mt], false, bfr[nt], (short)0, acc[mt][nt], false, false);
    }
    __syncthreads();
  }

  // Epilogue: C/D layout -> element e is row (e + 8*lh), col l16 within tile
#pragma unroll
  for (int nt = 0; nt < 4; nt++) {
    const int col = tile_n + wn + nt * 16 + l16;
    const float bv = bias[col];
#pragma unroll
    for (int mt = 0; mt < 2; mt++) {
      const int rbase = tile_m + wm + mt * 16 + lh * 8;
#pragma unroll
      for (int e = 0; e < 8; e++)
        out[(size_t)(rbase + e) * ADIM + col] = (__bf16)(acc[mt][nt][e] + bv);
    }
  }
}

// ---------------------------------------------------------------------------
// q projection: 4 rows only (last token per batch), full f32 precision.
// ---------------------------------------------------------------------------
__global__ __launch_bounds__(256) void qproj(const float* __restrict__ x,
                                             const float* __restrict__ Wq,
                                             const float* __restrict__ bq,
                                             float* __restrict__ qout) {
  const int idx = blockIdx.x * 256 + threadIdx.x;   // 0..4095
  const int b = idx >> 10;
  const int a = idx & (ADIM - 1);
  const float* xr = x + ((size_t)(b * SEQ + SEQ - 1)) * CDIM;
  const float* wr = Wq + (size_t)a * CDIM;
  float s = 0.f;
  for (int c = 0; c < CDIM; c += 4) {
    v4f xv = *(const v4f*)(xr + c);
    v4f wv = *(const v4f*)(wr + c);
    s += xv[0] * wv[0] + xv[1] * wv[1] + xv[2] * wv[2] + xv[3] * wv[3];
  }
  qout[idx] = s + bq[a];
}

// ---------------------------------------------------------------------------
// Flash-decoding partials: per (b,h,chunk of 1024 keys) compute local max m,
// l = sum exp(s-m), acc[d] = sum exp(s-m)*v. partials stride = 66 floats.
// ---------------------------------------------------------------------------
__global__ __launch_bounds__(256) void attn_partial(const __bf16* __restrict__ K,
                                                    const __bf16* __restrict__ V,
                                                    const float* __restrict__ q,
                                                    float* __restrict__ part) {
  const int S = 8, Tc = SEQ / S;  // 8 chunks x 1024
  const int tid = threadIdx.x;
  const int bh = blockIdx.x / S;
  const int chunk = blockIdx.x - bh * S;
  const int b = bh >> 4;
  const int h = bh & 15;

  __shared__ float qs[HDIM];
  __shared__ float ps[1024];
  __shared__ float red[256];
  __shared__ float pacc[4][HDIM];

  if (tid < HDIM) qs[tid] = q[b * ADIM + h * HDIM + tid];
  __syncthreads();

  const size_t krow0 = ((size_t)(b * SEQ + chunk * Tc)) * ADIM + h * HDIM;
  float sc[4];
  float smax = -1e30f;
#pragma unroll
  for (int j = 0; j < 4; j++) {
    const int t = j * 256 + tid;
    const v8bf* kr = (const v8bf*)(K + krow0 + (size_t)t * ADIM);
    float s = 0.f;
#pragma unroll
    for (int dd = 0; dd < 8; dd++) {
      v8bf kv = kr[dd];
#pragma unroll
      for (int e = 0; e < 8; e++) s += qs[dd * 8 + e] * (float)kv[e];
    }
    s *= 0.125f;  // 1/sqrt(HDIM)
    sc[j] = s;
    smax = fmaxf(smax, s);
  }
  red[tid] = smax;
  __syncthreads();
  for (int off = 128; off > 0; off >>= 1) {
    if (tid < off) red[tid] = fmaxf(red[tid], red[tid + off]);
    __syncthreads();
  }
  const float m = red[0];

  float lsum = 0.f;
#pragma unroll
  for (int j = 0; j < 4; j++) {
    const float p = __expf(sc[j] - m);
    ps[j * 256 + tid] = p;
    lsum += p;
  }
  __syncthreads();
  red[tid] = lsum;
  __syncthreads();
  for (int off = 128; off > 0; off >>= 1) {
    if (tid < off) red[tid] += red[tid + off];
    __syncthreads();
  }

  // weighted V sum: 4 thread groups x 64 dims, coalesced over d
  const int d = tid & (HDIM - 1);
  const int g = tid >> 6;
  const __bf16* vb = V + ((size_t)(b * SEQ + chunk * Tc + g * 256)) * ADIM + h * HDIM + d;
  float accv = 0.f;
  for (int t = 0; t < 256; t++) accv += ps[g * 256 + t] * (float)vb[(size_t)t * ADIM];
  pacc[g][d] = accv;
  __syncthreads();

  if (tid < HDIM) {
    float* pr = part + ((size_t)(bh * S + chunk)) * 66;
    pr[2 + tid] = pacc[0][tid] + pacc[1][tid] + pacc[2][tid] + pacc[3][tid];
    if (tid == 0) { pr[0] = m; pr[1] = red[0]; }
  }
}

// ---------------------------------------------------------------------------
// Merge the 8 chunk partials per (b,h) with log-sum-exp rescaling.
// ---------------------------------------------------------------------------
__global__ __launch_bounds__(64) void attn_combine(const float* __restrict__ part,
                                                   float* __restrict__ out) {
  const int S = 8;
  const int bh = blockIdx.x;
  const int d = threadIdx.x;
  const float* pr = part + (size_t)bh * S * 66;
  float M = -1e30f;
  for (int s = 0; s < S; s++) M = fmaxf(M, pr[s * 66]);
  float L = 0.f, y = 0.f;
  for (int s = 0; s < S; s++) {
    const float w = __expf(pr[s * 66] - M);
    L += pr[s * 66 + 1] * w;
    y += pr[s * 66 + 2 + d] * w;
  }
  out[bh * HDIM + d] = y / L;
}

extern "C" void kernel_launch(void* const* d_in, const int* in_sizes, int n_in,
                              void* d_out, int out_size, void* d_ws, size_t ws_size,
                              hipStream_t stream) {
  const float* x  = (const float*)d_in[0];
  const float* Wk = (const float*)d_in[1];
  const float* bk = (const float*)d_in[2];
  const float* Wq = (const float*)d_in[3];
  const float* bq = (const float*)d_in[4];
  const float* Wv = (const float*)d_in[5];
  const float* bv = (const float*)d_in[6];
  float* out = (float*)d_out;

  char* ws = (char*)d_ws;
  const size_t kvBytes = (size_t)MROWS * ADIM * sizeof(__bf16);  // 64 MB each
  __bf16* Kb   = (__bf16*)(ws);
  __bf16* Vb   = (__bf16*)(ws + kvBytes);
  float*  qb   = (float*)(ws + 2 * kvBytes);
  float*  prt  = (float*)(ws + 2 * kvBytes + (size_t)BATCH * ADIM * sizeof(float));

  dim3 gproj(ADIM / BN, MROWS / BM, 1);
  proj_gemm_bf16<<<gproj, 256, 0, stream>>>(x, Wk, bk, Kb);
  proj_gemm_bf16<<<gproj, 256, 0, stream>>>(x, Wv, bv, Vb);
  qproj<<<(BATCH * ADIM) / 256, 256, 0, stream>>>(x, Wq, bq, qb);
  attn_partial<<<BATCH * NHEAD * 8, 256, 0, stream>>>(Kb, Vb, qb, prt);
  attn_combine<<<BATCH * NHEAD, 64, 0, stream>>>(prt, out);
}
